// Attention_9706626089252
// MI455X (gfx1250) — matrix operand order repaired and checked by reading, not compiled, not run on hardware
//
#include <hip/hip_runtime.h>
#include <hip/hip_bf16.h>

typedef _Float16 v16h __attribute__((ext_vector_type(16)));
typedef _Float16 v8h  __attribute__((ext_vector_type(8)));
typedef float    v8f  __attribute__((ext_vector_type(8)));

#define B_     2
#define N_     2048
#define DIM_   2048
#define INNER_ 2048
#define DH_    128
#define H_     16
#define M_     4096   // B_*N_

// Async 16-byte global -> LDS copy (ASYNCcnt-tracked, bypasses VGPRs).
__device__ __forceinline__ void async_cp16(uint32_t lds_off, const void* gptr) {
    asm volatile("global_load_async_to_lds_b128 %0, %1, off"
                 :: "v"(lds_off), "v"((unsigned long long)(uintptr_t)gptr)
                 : "memory");
}
__device__ __forceinline__ uint32_t lds_off_u32(const void* p) {
    // flat LDS aperture address: low 32 bits are the LDS byte offset
    return (uint32_t)(uintptr_t)p;
}

// ---------------------------------------------------------------------------
// K0: transpose + cast f32 -> f16:  dst[c*R + r] = (f16) src[r*C + c]
// ---------------------------------------------------------------------------
__global__ __launch_bounds__(256) void transpose_cast(const float* __restrict__ src,
                                                      _Float16* __restrict__ dst,
                                                      int R, int C) {
    int idx = blockIdx.x * 256 + threadIdx.x;
    if (idx < R * C) {
        int r = idx / C, c = idx % C;
        dst[(size_t)c * R + r] = (_Float16)src[idx];
    }
}

// ---------------------------------------------------------------------------
// K1: RMSNorm (sum-of-squares form) + cast to f16. One block per row of 2048.
// ---------------------------------------------------------------------------
__global__ __launch_bounds__(256) void rmsnorm_kernel(const float* __restrict__ x,
                                                      const float* __restrict__ gamma,
                                                      _Float16* __restrict__ xn) {
    __shared__ float red[8];
    int row = blockIdx.x;
    int tid = threadIdx.x;
    const float* xr = x + (size_t)row * DIM_;
    float4 a  = ((const float4*)xr)[tid * 2 + 0];
    float4 b2 = ((const float4*)xr)[tid * 2 + 1];
    float s = a.x*a.x + a.y*a.y + a.z*a.z + a.w*a.w
            + b2.x*b2.x + b2.y*b2.y + b2.z*b2.z + b2.w*b2.w;
    #pragma unroll
    for (int off = 16; off > 0; off >>= 1) s += __shfl_xor(s, off, 32);
    if ((tid & 31) == 0) red[tid >> 5] = s;
    __syncthreads();
    float tot = 0.f;
    #pragma unroll
    for (int w = 0; w < 8; ++w) tot += red[w];
    float inv = rsqrtf(tot + 1e-5f) * 45.25483399593904f; // sqrt(2048)
    int off0 = tid * 8;
    const float* gp = gamma + off0;
    _Float16* op = xn + (size_t)row * DIM_ + off0;
    float v[8] = {a.x,a.y,a.z,a.w,b2.x,b2.y,b2.z,b2.w};
    #pragma unroll
    for (int i = 0; i < 8; ++i) op[i] = (_Float16)(v[i] * inv * gp[i]);
}

// ---------------------------------------------------------------------------
// K2: generic f16 WMMA GEMM:  C[M x N] = A[M x K] * BT[N x K]^T  (BT row-major)
// Block = 256 thr = 8 waves. Block tile 128(M) x 64(N); wave tile 16 x 64.
// BT k-slab staged into LDS via async copies (ASYNCcnt), shared by 8 waves.
// ---------------------------------------------------------------------------
template <typename OutT, bool TRANS>
__global__ __launch_bounds__(256) void gemm16(const _Float16* __restrict__ A,
                                              const _Float16* __restrict__ BT,
                                              OutT* __restrict__ C,
                                              int K, int lda, int ldb, int ldc,
                                              float scale) {
    __shared__ __align__(16) _Float16 lb[64 * 32];   // 64 cols x 32 k
    int tid  = threadIdx.x;
    int wave = tid >> 5, lane = tid & 31;
    int cc = lane & 15, hi = lane >> 4;
    int mblk = blockIdx.x * 128 + wave * 16;
    int nblk = blockIdx.y * 64;
    v8f acc[4] = {};
    const _Float16* arow = A + (size_t)(mblk + cc) * lda;
    int rrow = tid >> 2, part = tid & 3;
    const _Float16* bsrc = BT + (size_t)(nblk + rrow) * ldb + part * 8;
    uint32_t ldst = lds_off_u32(lb) + (uint32_t)tid * 16;

    for (int k0 = 0; k0 < K; k0 += 32) {
        __syncthreads();                       // WAR: everyone done with lb
        async_cp16(ldst, bsrc + k0);           // 256 thr x 16B = 4KB slab
        asm volatile("s_wait_asynccnt 0x0" ::: "memory");
        __syncthreads();                       // RAW: slab visible to all waves
        // A fragment (16x32): lanes 0-15 hold K {0..7,16..23}, 16-31 {8..15,24..31}
        v16h af;
        ((v8h*)&af)[0] = *(const v8h*)(arow + k0 + hi * 8);
        ((v8h*)&af)[1] = *(const v8h*)(arow + k0 + 16 + hi * 8);
        #pragma unroll
        for (int s = 0; s < 4; ++s) {
            // B fragment (32x16): lane holds one column, lanes 0-15 K {0..7,16..23}, 16-31 {8..15,24..31}
            const _Float16* bp = &lb[(s * 16 + cc) * 32 + hi * 8];
            v16h bf;
            ((v8h*)&bf)[0] = *(const v8h*)(bp);
            ((v8h*)&bf)[1] = *(const v8h*)(bp + 16);
            acc[s] = __builtin_amdgcn_wmma_f32_16x16x32_f16(
                false, af, false, bf, (short)0, acc[s], false, false);
        }
    }
    #pragma unroll
    for (int s = 0; s < 4; ++s) {
        #pragma unroll
        for (int e = 0; e < 8; ++e) {
            int m  = mblk + e + 8 * hi;
            int nn = nblk + s * 16 + cc;
            float val = acc[s][e] * scale;
            if (TRANS) C[(size_t)nn * ldc + m] = (OutT)val;
            else       C[(size_t)m  * ldc + nn] = (OutT)val;
        }
    }
}

// ---------------------------------------------------------------------------
// K3: flash attention core. Block = 4 waves = 4 consecutive q-tiles of the
// SAME (b,h) -> K/V shared. K and Vt slabs (16KB each) are double-buffered in
// LDS and filled with async copies so block j+1's HBM traffic overlaps block
// j's WMMA + softmax. s_wait_asynccnt 0x10 retires exactly the current
// buffer (async loads complete in order; 16/thread in flight = next buffer).
// ---------------------------------------------------------------------------
__global__ __launch_bounds__(128) void attn_kernel(const _Float16* __restrict__ q,
                                                   const _Float16* __restrict__ k,
                                                   const _Float16* __restrict__ vT,
                                                   const float* __restrict__ bias,
                                                   _Float16* __restrict__ aout) {
    __shared__ __align__(16) _Float16 ldsK[2][64 * 128];  // [j-local][d]
    __shared__ __align__(16) _Float16 ldsV[2][128 * 64];  // [d][j-local]
    __shared__ __align__(16) _Float16 ldsP[4 * 16 * 64];
    int tid = threadIdx.x;
    int wave = tid >> 5, lane = tid & 31;
    int cc = lane & 15, hi = lane >> 4;
    int tile = blockIdx.x * 4 + wave;
    int qt = tile & 127;
    int h  = (tile >> 7) & 15;
    int b  = tile >> 11;
    int i0 = qt * 16;

    // staging sources (per-thread): K slab is 16KB contiguous; V row = d = tid
    const char* ksrc = (const char*)(k + (size_t)b * N_ * DH_);
    const char* vsrc = (const char*)(vT + (size_t)tid * M_ + (size_t)b * N_);
    uint32_t kdst0 = lds_off_u32(&ldsK[0][0]) + (uint32_t)tid * 128;
    uint32_t kdst1 = lds_off_u32(&ldsK[1][0]) + (uint32_t)tid * 128;
    uint32_t vdst0 = lds_off_u32(&ldsV[0][0]) + (uint32_t)tid * 128;
    uint32_t vdst1 = lds_off_u32(&ldsV[1][0]) + (uint32_t)tid * 128;
    auto stage = [&](int buf, int j0) {
        uint32_t kd = buf ? kdst1 : kdst0;
        const char* kp = ksrc + (size_t)j0 * (DH_ * 2) + (size_t)tid * 128;
        #pragma unroll
        for (int i = 0; i < 8; ++i) async_cp16(kd + i * 16, kp + i * 16);
        uint32_t vd = buf ? vdst1 : vdst0;
        const char* vp = vsrc + (size_t)j0 * 2;
        #pragma unroll
        for (int i = 0; i < 8; ++i) async_cp16(vd + i * 16, vp + i * 16);
    };

    // q fragments for the whole 128-wide head dim (4 x K=32)
    const _Float16* qb = q + (size_t)(b * N_ + i0 + cc) * INNER_ + h * DH_;
    v16h qf[4];
    #pragma unroll
    for (int kk = 0; kk < 4; ++kk) {
        ((v8h*)&qf[kk])[0] = *(const v8h*)(qb + kk * 32 + hi * 8);
        ((v8h*)&qf[kk])[1] = *(const v8h*)(qb + kk * 32 + 16 + hi * 8);
    }

    v8f O[8] = {};
    float m[8], l[8];
    #pragma unroll
    for (int e = 0; e < 8; ++e) { m[e] = -3.0e38f; l[e] = 0.f; }
    _Float16* lp = ldsP + wave * (16 * 64);
    const float* brow = bias + ((size_t)h * N_ + (i0 + 8 * hi)) * N_ + cc;

    stage(0, 0);                                        // prologue: buffer 0
    for (int jb = 0; jb < N_ / 64; ++jb) {
        int cur = jb & 1;
        int j0 = jb * 64;
        if (jb + 1 < N_ / 64) {
            stage(1 - cur, j0 + 64);                    // overlap next block
            asm volatile("s_wait_asynccnt 0x10" ::: "memory");
            // prefetch next bias block while WMMAs run
            const float* nb = brow + j0 + 64;
            #pragma unroll
            for (int e = 0; e < 8; ++e) __builtin_prefetch(nb + (size_t)e * N_, 0, 0);
        } else {
            asm volatile("s_wait_asynccnt 0x0" ::: "memory");
        }
        __syncthreads();                                // cur slab visible

        v8f S[4];
        #pragma unroll
        for (int t = 0; t < 4; ++t) {
            v8f s = {};
            const _Float16* kb = &ldsK[cur][(t * 16 + cc) * 128];
            #pragma unroll
            for (int kk = 0; kk < 4; ++kk) {
                v16h bf;
                ((v8h*)&bf)[0] = *(const v8h*)(kb + kk * 32 + hi * 8);
                ((v8h*)&bf)[1] = *(const v8h*)(kb + kk * 32 + hi * 8 + 16);
                s = __builtin_amdgcn_wmma_f32_16x16x32_f16(
                    false, qf[kk], false, bf, (short)0, s, false, false);
            }
            const float* bp = brow + j0 + t * 16;
            #pragma unroll
            for (int e = 0; e < 8; ++e) s[e] += bp[(size_t)e * N_];
            S[t] = s;
        }
        // online softmax: row max / rescale / exp / row sum
        float alpha[8];
        #pragma unroll
        for (int e = 0; e < 8; ++e) {
            float mx = fmaxf(fmaxf(S[0][e], S[1][e]), fmaxf(S[2][e], S[3][e]));
            #pragma unroll
            for (int off = 1; off < 16; off <<= 1) mx = fmaxf(mx, __shfl_xor(mx, off, 16));
            float mn = fmaxf(m[e], mx);
            alpha[e] = __expf(m[e] - mn);
            m[e] = mn;
        }
        #pragma unroll
        for (int d = 0; d < 8; ++d)
            #pragma unroll
            for (int e = 0; e < 8; ++e) O[d][e] *= alpha[e];
        #pragma unroll
        for (int e = 0; e < 8; ++e) {
            float sm = 0.f;
            #pragma unroll
            for (int t = 0; t < 4; ++t) {
                float p = __expf(S[t][e] - m[e]);
                S[t][e] = p;
                sm += p;
            }
            #pragma unroll
            for (int off = 1; off < 16; off <<= 1) sm += __shfl_xor(sm, off, 16);
            l[e] = l[e] * alpha[e] + sm;
        }
        // P: C-layout -> f16 -> LDS -> A-layout fragments (same-wave, DS in-order)
        #pragma unroll
        for (int t = 0; t < 4; ++t)
            #pragma unroll
            for (int e = 0; e < 8; ++e)
                lp[(e + 8 * hi) * 64 + t * 16 + cc] = (_Float16)S[t][e];
        asm volatile("s_wait_dscnt 0x0" ::: "memory");
        #pragma unroll
        for (int kk = 0; kk < 2; ++kk) {
            v16h pf;
            ((v8h*)&pf)[0] = *(const v8h*)(lp + cc * 64 + kk * 32 + hi * 8);
            ((v8h*)&pf)[1] = *(const v8h*)(lp + cc * 64 + kk * 32 + 16 + hi * 8);
            #pragma unroll
            for (int d8 = 0; d8 < 8; ++d8) {
                const _Float16* vb = &ldsV[cur][(d8 * 16 + cc) * 64 + kk * 32 + hi * 8];
                v16h bf;
                ((v8h*)&bf)[0] = *(const v8h*)(vb);
                ((v8h*)&bf)[1] = *(const v8h*)(vb + 16);
                O[d8] = __builtin_amdgcn_wmma_f32_16x16x32_f16(
                    false, pf, false, bf, (short)0, O[d8], false, false);
            }
        }
        __syncthreads();                // all waves done reading cur slab (WAR)
    }
    // normalize + store attn output (b, n, h*dh) in f16
    #pragma unroll
    for (int d8 = 0; d8 < 8; ++d8)
        #pragma unroll
        for (int e = 0; e < 8; ++e) {
            size_t row = (size_t)(b * N_ + i0 + e + 8 * hi);
            aout[row * INNER_ + h * DH_ + d8 * 16 + cc] = (_Float16)(O[d8][e] / l[e]);
        }
}

// ---------------------------------------------------------------------------
extern "C" void kernel_launch(void* const* d_in, const int* in_sizes, int n_in,
                              void* d_out, int out_size, void* d_ws, size_t ws_size,
                              hipStream_t stream) {
    (void)in_sizes; (void)n_in; (void)out_size; (void)ws_size;
    const float* x     = (const float*)d_in[0];
    const float* bias  = (const float*)d_in[1];
    const float* gamma = (const float*)d_in[2];
    const float* wq    = (const float*)d_in[3];
    const float* wk    = (const float*)d_in[4];
    const float* wv    = (const float*)d_in[5];
    const float* wo    = (const float*)d_in[6];
    float* out = (float*)d_out;

    char* ws = (char*)d_ws;
    size_t off = 0;
    auto alloc = [&](size_t bytes) { void* p = ws + off; off = (off + bytes + 255) & ~(size_t)255; return p; };
    _Float16* xn   = (_Float16*)alloc((size_t)M_ * DIM_ * 2);
    _Float16* wqT  = (_Float16*)alloc((size_t)INNER_ * DIM_ * 2);
    _Float16* wkT  = (_Float16*)alloc((size_t)DH_ * DIM_ * 2);
    _Float16* wvT  = (_Float16*)alloc((size_t)DH_ * DIM_ * 2);
    _Float16* woT  = (_Float16*)alloc((size_t)DIM_ * INNER_ * 2);
    _Float16* qbuf = (_Float16*)alloc((size_t)M_ * INNER_ * 2);
    _Float16* kbuf = (_Float16*)alloc((size_t)M_ * DH_ * 2);
    _Float16* vTb  = (_Float16*)alloc((size_t)DH_ * M_ * 2);
    _Float16* aout = (_Float16*)alloc((size_t)M_ * INNER_ * 2);

    // weight prep (f32 -> f16, transposed for contiguous B-fragment loads)
    transpose_cast<<<(DIM_ * INNER_ + 255) / 256, 256, 0, stream>>>(wq, wqT, DIM_, INNER_);
    transpose_cast<<<(DIM_ * DH_    + 255) / 256, 256, 0, stream>>>(wk, wkT, DIM_, DH_);
    transpose_cast<<<(DIM_ * DH_    + 255) / 256, 256, 0, stream>>>(wv, wvT, DIM_, DH_);
    transpose_cast<<<(INNER_ * DIM_ + 255) / 256, 256, 0, stream>>>(wo, woT, INNER_, DIM_);

    // RMSNorm
    rmsnorm_kernel<<<M_, 256, 0, stream>>>(x, gamma, xn);

    // projections (q folds the 1/sqrt(128) scale; v is written transposed)
    gemm16<_Float16, false><<<dim3(M_ / 128, INNER_ / 64), 256, 0, stream>>>(
        xn, wqT, qbuf, DIM_, DIM_, DIM_, INNER_, 0.08838834764831845f);
    gemm16<_Float16, false><<<dim3(M_ / 128, DH_ / 64), 256, 0, stream>>>(
        xn, wkT, kbuf, DIM_, DIM_, DIM_, DH_, 1.0f);
    gemm16<_Float16, true><<<dim3(M_ / 128, DH_ / 64), 256, 0, stream>>>(
        xn, wvT, vTb, DIM_, DIM_, DIM_, M_, 1.0f);

    // flash attention core: 4096 q-tiles, 4 waves (same b,h) per block
    attn_kernel<<<M_ * H_ / 16 / 4, 128, 0, stream>>>(qbuf, kbuf, vTb, bias, aout);

    // output projection -> f32
    gemm16<float, false><<<dim3(M_ / 128, DIM_ / 64), 256, 0, stream>>>(
        aout, woT, out, INNER_, INNER_, INNER_, DIM_, 1.0f);
}
